// LSTM_32512902431037
// MI455X (gfx1250) — compile-verified, matrix operand order checked by reading, not run on hardware
//
#include <hip/hip_runtime.h>
#include <hip/hip_bf16.h>
#include <math.h>
#include <stdint.h>

typedef __attribute__((ext_vector_type(16))) __bf16 v16bf;
typedef __attribute__((ext_vector_type(8)))  float  v8f;

namespace {
constexpr int kH   = 516;    // hidden per direction
constexpr int kHp  = 544;    // K-padded hidden (mult of 32)
constexpr int kG   = 2064;   // 4H (gates)
constexpr int kGp  = 2080;   // padded 4H (mult of 32/16)
constexpr int kDH  = 1032;   // 2H
constexpr int kDHK = 1056;   // K-padded 2H
constexpr int kDHN = 1040;   // N-padded 2H
constexpr int kT   = 512;
constexpr int kB   = 64;
constexpr int kTB  = kT * kB;
constexpr int kNY  = 20;
constexpr int kSEQ = 20;
}

// ---- WMMA fragment loader: 16x32 bf16 tile, row-major source -------------
// A-layout (ISA 7.12.2): lane&15 = row; lane>>4 selects K halves {0..7,16..23}
// vs {8..15,24..31}; two bf16 per dword -> dwords [hi*4+d] and [8+hi*4+d].
__device__ inline v16bf load_frag(const __bf16* p, int ld) {
  const int lane = threadIdx.x & 31;
  const int row  = lane & 15;
  const int hi   = lane >> 4;
  const uint32_t* r32 = (const uint32_t*)(p + (size_t)row * ld);
  union { v16bf v; uint32_t u[8]; } t;
#pragma unroll
  for (int d = 0; d < 4; ++d) t.u[d] = r32[hi * 4 + d];
#pragma unroll
  for (int d = 0; d < 4; ++d) t.u[4 + d] = r32[8 + hi * 4 + d];
  return t.v;
}

// ---- Generic GEMM: C[M,N] = A[M,Kp] * W[N,Kp]^T (+bias) (+=) -------------
// One wave per 16x16 tile, 8 waves per block. f32 or bf16 output.
__global__ void gemm_bf16(const __bf16* __restrict__ A, int lda,
                          const __bf16* __restrict__ W, int ldw,
                          float* __restrict__ Cf, __bf16* __restrict__ Cb, int ldc,
                          const float* __restrict__ bias, int nBias,
                          int mTiles, int nTiles, int kIters, int accumulate) {
  const int wave = threadIdx.x >> 5;
  int tile = blockIdx.x * (blockDim.x >> 5) + wave;
  if (tile >= mTiles * nTiles) return;
  const int mt = tile / nTiles, nt = tile % nTiles;
  const __bf16* Ap = A + (size_t)(mt * 16) * lda;
  const __bf16* Wp = W + (size_t)(nt * 16) * ldw;
  v8f acc = {};
  for (int k = 0; k < kIters; ++k) {
    v16bf a = load_frag(Ap + k * 32, lda);
    v16bf b = load_frag(Wp + k * 32, ldw);
    acc = __builtin_amdgcn_wmma_f32_16x16x32_bf16(false, a, false, b,
                                                  (short)0, acc, false, false);
  }
  const int lane = threadIdx.x & 31;
  const int n  = nt * 16 + (lane & 15);
  const int m0 = mt * 16 + ((lane >> 4) * 8);
  const float bv = (bias != nullptr && n < nBias) ? bias[n] : 0.f;
#pragma unroll
  for (int r = 0; r < 8; ++r) {
    const size_t o = (size_t)(m0 + r) * ldc + n;
    float v = acc[r] + bv;
    if (Cb) {
      Cb[o] = (__bf16)v;
    } else {
      if (accumulate) v += Cf[o];
      Cf[o] = v;
    }
  }
}

// ---- Fused LSTM step GEMM: gates = x@Wi^T + h@Wh^T + b, both dirs via z --
__global__ void lstm_step_gemm(const __bf16* __restrict__ x0,
                               const __bf16* __restrict__ x1,
                               int lda1, int k1Iters,
                               const __bf16* __restrict__ Wi,
                               const __bf16* __restrict__ Wh,
                               const __bf16* __restrict__ h,
                               const float* __restrict__ bias,
                               float* __restrict__ gates) {
  const int dir  = blockIdx.z;
  const int wave = threadIdx.x >> 5;
  const int nTiles = kGp / 16;                 // 130
  int tile = blockIdx.x * (blockDim.x >> 5) + wave;
  if (tile >= 4 * nTiles) return;              // M=64 -> 4 row tiles
  const int mt = tile / nTiles, nt = tile % nTiles;
  const __bf16* x   = (dir ? x1 : x0) + (size_t)(mt * 16) * lda1;
  const __bf16* Wd  = Wi + (size_t)dir * kGp * lda1 + (size_t)(nt * 16) * lda1;
  const __bf16* hd  = h  + (size_t)dir * kB * kHp  + (size_t)(mt * 16) * kHp;
  const __bf16* Whd = Wh + (size_t)dir * kGp * kHp + (size_t)(nt * 16) * kHp;
  v8f acc = {};
  for (int k = 0; k < k1Iters; ++k) {
    v16bf a = load_frag(x + k * 32, lda1);
    v16bf b = load_frag(Wd + k * 32, lda1);
    acc = __builtin_amdgcn_wmma_f32_16x16x32_bf16(false, a, false, b,
                                                  (short)0, acc, false, false);
  }
#pragma unroll 1
  for (int k = 0; k < kHp / 32; ++k) {         // 17 recurrent K-steps
    v16bf a = load_frag(hd + k * 32, kHp);
    v16bf b = load_frag(Whd + k * 32, kHp);
    acc = __builtin_amdgcn_wmma_f32_16x16x32_bf16(false, a, false, b,
                                                  (short)0, acc, false, false);
  }
  const int lane = threadIdx.x & 31;
  const int n  = nt * 16 + (lane & 15);
  const int m0 = mt * 16 + ((lane >> 4) * 8);
  const float bv = (n < kG) ? bias[dir * kG + n] : 0.f;
  float* gd = gates + (size_t)dir * kB * kGp;
#pragma unroll
  for (int r = 0; r < 8; ++r) gd[(size_t)(m0 + r) * kGp + n] = acc[r] + bv;
}

// ---- LSTM cell elementwise (encoder: writes into time-sequence buffer) ---
__global__ void enc_cell(const float* __restrict__ gates, float* __restrict__ c,
                         __bf16* __restrict__ h, __bf16* __restrict__ seq,
                         int t_f, int t_b) {
  int idx = blockIdx.x * blockDim.x + threadIdx.x;
  if (idx >= 2 * kB * kHp) return;
  const int d = idx / (kB * kHp);
  const int rem = idx - d * kB * kHp;
  const int b = rem / kHp, cc = rem % kHp;
  if (cc >= kH) { h[idx] = (__bf16)0.f; return; }
  const float* g = gates + (size_t)d * kB * kGp + (size_t)b * kGp;
  const float si = 1.f / (1.f + __expf(-g[cc]));
  const float sf = 1.f / (1.f + __expf(-g[kH + cc]));
  const float gg = tanhf(g[2 * kH + cc]);
  const float so = 1.f / (1.f + __expf(-g[3 * kH + cc]));
  float* cp = c + (size_t)d * kB * kH + (size_t)b * kH + cc;
  const float cn = sf * (*cp) + si * gg;
  *cp = cn;
  const float hn = so * tanhf(cn);
  h[idx] = (__bf16)hn;
  const int t = d ? t_b : t_f;
  seq[(size_t)(t * kB + b) * kDHK + d * kH + cc] = (__bf16)hn;
}

// ---- LSTM cell elementwise (decoder: writes into [B, 2H] concat buffer) --
__global__ void dec_cell(const float* __restrict__ gates, float* __restrict__ c,
                         __bf16* __restrict__ h, __bf16* __restrict__ outcat) {
  int idx = blockIdx.x * blockDim.x + threadIdx.x;
  if (idx >= 2 * kB * kHp) return;
  const int d = idx / (kB * kHp);
  const int rem = idx - d * kB * kHp;
  const int b = rem / kHp, cc = rem % kHp;
  if (cc >= kH) { h[idx] = (__bf16)0.f; return; }
  const float* g = gates + (size_t)d * kB * kGp + (size_t)b * kGp;
  const float si = 1.f / (1.f + __expf(-g[cc]));
  const float sf = 1.f / (1.f + __expf(-g[kH + cc]));
  const float gg = tanhf(g[2 * kH + cc]);
  const float so = 1.f / (1.f + __expf(-g[3 * kH + cc]));
  float* cp = c + (size_t)d * kB * kH + (size_t)b * kH + cc;
  const float cn = sf * (*cp) + si * gg;
  *cp = cn;
  const float hn = so * tanhf(cn);
  h[idx] = (__bf16)hn;
  outcat[(size_t)b * kDHK + d * kH + cc] = (__bf16)hn;
}

// ---- score[t,b] = sum_h v[h] * tanh(hidPart[b,h] + encPart[t*B+b,h]) -----
__global__ void score_k(const float* __restrict__ hidPart,
                        const __bf16* __restrict__ encPart,
                        const float* __restrict__ v, float* __restrict__ score) {
  const int wave = threadIdx.x >> 5, lane = threadIdx.x & 31;
  const int gid = blockIdx.x * 8 + wave;
  if (gid >= kTB) return;
  const int t = gid / kB, b = gid % kB;
  const float* hp = hidPart + (size_t)b * kDHN;
  const __bf16* ep = encPart + (size_t)(t * kB + b) * kDHN;
  float s = 0.f;
  for (int h = lane; h < kDH; h += 32)
    s += v[h] * tanhf(hp[h] + (float)ep[h]);
#pragma unroll
  for (int off = 16; off; off >>= 1) s += __shfl_down(s, off);
  if (lane == 0) score[t * kB + b] = s;
}

// ---- softmax over T for each batch column --------------------------------
__global__ void softmax_k(const float* __restrict__ score, float* __restrict__ aw) {
  const int b = blockIdx.x, tid = threadIdx.x;
  __shared__ float red[256];
  float m = -1e30f;
  for (int t = tid; t < kT; t += 256) m = fmaxf(m, score[t * kB + b]);
  red[tid] = m; __syncthreads();
  for (int s = 128; s; s >>= 1) { if (tid < s) red[tid] = fmaxf(red[tid], red[tid + s]); __syncthreads(); }
  m = red[0]; __syncthreads();
  float sum = 0.f;
  for (int t = tid; t < kT; t += 256) sum += __expf(score[t * kB + b] - m);
  red[tid] = sum; __syncthreads();
  for (int s = 128; s; s >>= 1) { if (tid < s) red[tid] += red[tid + s]; __syncthreads(); }
  const float inv = 1.f / red[0];
  for (int t = tid; t < kT; t += 256)
    aw[t * kB + b] = __expf(score[t * kB + b] - m) * inv;
}

// ---- context[b,h] = sum_t aw[t,b] * outenc[t,b,h]  (bf16 out, L2-resident)
__global__ void context_k(const float* __restrict__ aw,
                          const __bf16* __restrict__ outenc,
                          __bf16* __restrict__ ctx) {
  int idx = blockIdx.x * blockDim.x + threadIdx.x;
  if (idx >= kB * kDHK) return;
  const int b = idx / kDHK, h = idx % kDHK;
  float s = 0.f;
  for (int t = 0; t < kT; ++t)
    s += aw[t * kB + b] * (float)outenc[(size_t)(t * kB + b) * kDHK + h];
  ctx[idx] = (__bf16)s;
}

// ---- co = tanh(co_tmp) -> bf16 padded ------------------------------------
__global__ void tanhcvt_k(const float* __restrict__ src, __bf16* __restrict__ dst) {
  int idx = blockIdx.x * blockDim.x + threadIdx.x;
  if (idx >= kB * kDHK) return;
  const int b = idx / kDHK, h = idx % kDHK;
  dst[idx] = (h < kDHN) ? (__bf16)tanhf(src[(size_t)b * kDHN + h]) : (__bf16)0.f;
}

// ---- cross-entropy for one decoder step, mean over batch, += into loss ---
__global__ void ce_k(const float* __restrict__ logits, const int* __restrict__ label,
                     float* __restrict__ out) {
  __shared__ float red[64];
  const int b = threadIdx.x;
  const float* l = logits + (size_t)b * 32;
  float m = -1e30f;
  for (int j = 0; j < kSEQ; ++j) m = fmaxf(m, l[j]);
  float s = 0.f;
  for (int j = 0; j < kSEQ; ++j) s += __expf(l[j] - m);
  const float lse = m + __logf(s);
  red[b] = lse - l[label[b]];
  __syncthreads();
  for (int st = 32; st; st >>= 1) { if (b < st) red[b] += red[b + st]; __syncthreads(); }
  if (b == 0) atomicAdd(out, red[0] * (1.0f / kB));
}

// ---- utility: f32 -> zero-padded bf16 [rp, cp] ---------------------------
__global__ void cvt_pad(const float* __restrict__ src, int sstride, int scoloff,
                        __bf16* __restrict__ dst, int rows, int cols, int cp, int total) {
  int i = blockIdx.x * blockDim.x + threadIdx.x;
  if (i >= total) return;
  const int r = i / cp, c = i - r * cp;
  const float v = (r < rows && c < cols) ? src[(size_t)r * sstride + scoloff + c] : 0.f;
  dst[i] = (__bf16)v;
}

__global__ void fill_zero(uint32_t* p, size_t n) {
  size_t i = (size_t)blockIdx.x * blockDim.x + threadIdx.x;
  if (i < n) p[i] = 0u;
}
__global__ void set_zero1(float* p) { if (threadIdx.x == 0 && blockIdx.x == 0) *p = 0.f; }

// ==========================================================================
extern "C" void kernel_launch(void* const* d_in, const int* in_sizes, int n_in,
                              void* d_out, int out_size, void* d_ws, size_t ws_size,
                              hipStream_t stream) {
  const float* x        = (const float*)d_in[0];
  const float* y        = (const float*)d_in[1];
  const int*   label    = (const int*)d_in[2];
  const float* expand_W = (const float*)d_in[3];
  const float* expand_b = (const float*)d_in[4];
  const float* eWi0     = (const float*)d_in[5];
  const float* eWh0     = (const float*)d_in[6];
  const float* eb0      = (const float*)d_in[7];
  const float* eWi1     = (const float*)d_in[8];
  const float* eWh1     = (const float*)d_in[9];
  const float* eb1      = (const float*)d_in[10];
  const float* dWi0     = (const float*)d_in[11];
  const float* dWh0     = (const float*)d_in[12];
  const float* db0      = (const float*)d_in[13];
  const float* dWi1     = (const float*)d_in[14];
  const float* dWh1     = (const float*)d_in[15];
  const float* db1      = (const float*)d_in[16];
  const float* attn_W   = (const float*)d_in[17];
  const float* attn_b   = (const float*)d_in[18];
  const float* attn_v   = (const float*)d_in[19];
  const float* concat_W = (const float*)d_in[20];
  const float* concat_b = (const float*)d_in[21];
  const float* out_W    = (const float*)d_in[22];
  const float* out_b    = (const float*)d_in[23];
  float* loss = (float*)d_out;

  // ---- workspace carve-up ----
  char* base = (char*)d_ws; size_t off = 0;
  auto alloc = [&](size_t bytes) -> void* {
    void* p = base + off; off = (off + bytes + 255) & ~(size_t)255; return p;
  };
  __bf16* x_bf    = (__bf16*)alloc((size_t)kTB * 32 * 2);
  __bf16* xe_bf   = (__bf16*)alloc((size_t)kTB * 256 * 2);
  __bf16* y_bf    = (__bf16*)alloc((size_t)kNY * kB * 32 * 2);
  __bf16* seq0    = (__bf16*)alloc((size_t)kTB * kDHK * 2);
  __bf16* outenc  = (__bf16*)alloc((size_t)kTB * kDHK * 2);
  __bf16* encPart = (__bf16*)alloc((size_t)kTB * kDHN * 2);
  float*  score   = (float*)alloc((size_t)kTB * 4);
  float*  aw      = (float*)alloc((size_t)kTB * 4);
  float*  gates   = (float*)alloc((size_t)2 * kB * kGp * 4);
  __bf16* h_enc   = (__bf16*)alloc((size_t)2 * kB * kHp * 2);
  float*  c_enc   = (float*)alloc((size_t)2 * kB * kH * 4);
  __bf16* dh      = (__bf16*)alloc((size_t)4 * kB * kHp * 2);
  float*  dc      = (float*)alloc((size_t)4 * kB * kH * 4);
  __bf16* l0      = (__bf16*)alloc((size_t)kB * kDHK * 2);
  __bf16* outdec  = (__bf16*)alloc((size_t)kB * kDHK * 2);
  float*  hidPart = (float*)alloc((size_t)kB * kDHN * 4);
  __bf16* ctx     = (__bf16*)alloc((size_t)kB * kDHK * 2);
  float*  co_tmp  = (float*)alloc((size_t)kB * kDHN * 4);
  __bf16* co_bf   = (__bf16*)alloc((size_t)kB * kDHK * 2);
  float*  logits  = (float*)alloc((size_t)kB * 32 * 4);
  __bf16* expandW_bf = (__bf16*)alloc((size_t)256 * 32 * 2);
  __bf16* eWi0_bf = (__bf16*)alloc((size_t)2 * kGp * 256 * 2);
  __bf16* eWh0_bf = (__bf16*)alloc((size_t)2 * kGp * kHp * 2);
  __bf16* eWi1_bf = (__bf16*)alloc((size_t)2 * kGp * kDHK * 2);
  __bf16* eWh1_bf = (__bf16*)alloc((size_t)2 * kGp * kHp * 2);
  __bf16* dWi0_bf = (__bf16*)alloc((size_t)2 * kGp * 32 * 2);
  __bf16* dWh0_bf = (__bf16*)alloc((size_t)2 * kGp * kHp * 2);
  __bf16* dWi1_bf = (__bf16*)alloc((size_t)2 * kGp * kDHK * 2);
  __bf16* dWh1_bf = (__bf16*)alloc((size_t)2 * kGp * kHp * 2);
  __bf16* attnW1_bf   = (__bf16*)alloc((size_t)kDHN * kDHK * 2);
  __bf16* attnW2_bf   = (__bf16*)alloc((size_t)kDHN * kDHK * 2);
  __bf16* concatW1_bf = (__bf16*)alloc((size_t)kDHN * kDHK * 2);
  __bf16* concatW2_bf = (__bf16*)alloc((size_t)kDHN * kDHK * 2);
  __bf16* outW_bf     = (__bf16*)alloc((size_t)32 * kDHK * 2);
  (void)ws_size; (void)n_in; (void)in_sizes; (void)out_size;

  auto cvt = [&](const float* src, int ss, int co, __bf16* dst, int rows, int cols,
                 int rp, int cp) {
    int total = rp * cp;
    cvt_pad<<<dim3((total + 255) / 256), dim3(256), 0, stream>>>(src, ss, co, dst,
                                                                 rows, cols, cp, total);
  };
  auto zero = [&](void* p, size_t bytes) {
    size_t n = bytes / 4;
    fill_zero<<<dim3((unsigned)((n + 255) / 256)), dim3(256), 0, stream>>>((uint32_t*)p, n);
  };
  auto gemm = [&](const __bf16* A, int lda, const __bf16* W, int ldw, float* Cf,
                  __bf16* Cb, int ldc, const float* bias, int nBias, int mT, int nT,
                  int kIt, int acc) {
    int tiles = mT * nT;
    gemm_bf16<<<dim3((tiles + 7) / 8), dim3(256), 0, stream>>>(A, lda, W, ldw, Cf, Cb,
                                                               ldc, bias, nBias, mT, nT,
                                                               kIt, acc);
  };

  set_zero1<<<1, 32, 0, stream>>>(loss);

  // ---- weight / input conversions to padded bf16 ----
  cvt(x, 4, 0, x_bf, kTB, 4, kTB, 32);
  cvt(y, kSEQ, 0, y_bf, kNY * kB, kSEQ, kNY * kB, 32);
  cvt(expand_W, 4, 0, expandW_bf, 256, 4, 256, 32);
  for (int d = 0; d < 2; ++d) {
    cvt(eWi0 + (size_t)d * kG * 256, 256, 0, eWi0_bf + (size_t)d * kGp * 256, kG, 256, kGp, 256);
    cvt(eWh0 + (size_t)d * kG * kH, kH, 0, eWh0_bf + (size_t)d * kGp * kHp, kG, kH, kGp, kHp);
    cvt(eWi1 + (size_t)d * kG * kDH, kDH, 0, eWi1_bf + (size_t)d * kGp * kDHK, kG, kDH, kGp, kDHK);
    cvt(eWh1 + (size_t)d * kG * kH, kH, 0, eWh1_bf + (size_t)d * kGp * kHp, kG, kH, kGp, kHp);
    cvt(dWi0 + (size_t)d * kG * kSEQ, kSEQ, 0, dWi0_bf + (size_t)d * kGp * 32, kG, kSEQ, kGp, 32);
    cvt(dWh0 + (size_t)d * kG * kH, kH, 0, dWh0_bf + (size_t)d * kGp * kHp, kG, kH, kGp, kHp);
    cvt(dWi1 + (size_t)d * kG * kDH, kDH, 0, dWi1_bf + (size_t)d * kGp * kDHK, kG, kDH, kGp, kDHK);
    cvt(dWh1 + (size_t)d * kG * kH, kH, 0, dWh1_bf + (size_t)d * kGp * kHp, kG, kH, kGp, kHp);
  }
  cvt(attn_W, 2 * kDH, 0,   attnW1_bf, kDH, kDH, kDHN, kDHK);
  cvt(attn_W, 2 * kDH, kDH, attnW2_bf, kDH, kDH, kDHN, kDHK);
  cvt(concat_W, 2 * kDH, 0,   concatW1_bf, kDH, kDH, kDHN, kDHK);
  cvt(concat_W, 2 * kDH, kDH, concatW2_bf, kDH, kDH, kDHN, kDHK);
  cvt(out_W, kDH, 0, outW_bf, kSEQ, kDH, 32, kDHK);

  // pad regions of sequence/concat buffers must be zero for WMMA K-padding
  zero(seq0, (size_t)kTB * kDHK * 2);
  zero(outenc, (size_t)kTB * kDHK * 2);
  zero(l0, (size_t)kB * kDHK * 2);
  zero(outdec, (size_t)kB * kDHK * 2);

  // ---- xe = x @ expand_W^T + expand_b  -> bf16 [TB,256] ----
  gemm(x_bf, 32, expandW_bf, 32, nullptr, xe_bf, 256, expand_b, 256, kTB / 16, 16, 1, 0);

  // ---- encoder biLSTM (2 layers) ----
  const int cellBlocks = (2 * kB * kHp + 255) / 256;
  auto run_layer = [&](const __bf16* inSeq, int lda1, int k1It, const __bf16* Wi,
                       const __bf16* Wh, const float* bias2, __bf16* outSeq) {
    zero(h_enc, (size_t)2 * kB * kHp * 2);
    zero(c_enc, (size_t)2 * kB * kH * 4);
    for (int t = 0; t < kT; ++t) {
      const int tf = t, tb = kT - 1 - t;
      lstm_step_gemm<<<dim3(65, 1, 2), dim3(256), 0, stream>>>(
          inSeq + (size_t)tf * kB * lda1, inSeq + (size_t)tb * kB * lda1, lda1, k1It,
          Wi, Wh, h_enc, bias2, gates);
      enc_cell<<<dim3(cellBlocks), dim3(256), 0, stream>>>(gates, c_enc, h_enc, outSeq,
                                                           tf, tb);
    }
  };
  run_layer(xe_bf, 256, 256 / 32, eWi0_bf, eWh0_bf, eb0, seq0);
  hipMemcpyAsync(dh, h_enc, (size_t)2 * kB * kHp * 2, hipMemcpyDeviceToDevice, stream);
  hipMemcpyAsync(dc, c_enc, (size_t)2 * kB * kH * 4, hipMemcpyDeviceToDevice, stream);
  run_layer(seq0, kDHK, kDHK / 32, eWi1_bf, eWh1_bf, eb1, outenc);
  hipMemcpyAsync(dh + (size_t)2 * kB * kHp, h_enc, (size_t)2 * kB * kHp * 2,
                 hipMemcpyDeviceToDevice, stream);
  hipMemcpyAsync(dc + (size_t)2 * kB * kH, c_enc, (size_t)2 * kB * kH * 4,
                 hipMemcpyDeviceToDevice, stream);

  // ---- attention precompute: encPart = outenc @ attn_W[:,1032:]^T + attn_b
  gemm(outenc, kDHK, attnW2_bf, kDHK, nullptr, encPart, kDHN, attn_b, kDH,
       kTB / 16, kDHN / 16, kDHK / 32, 0);

  // ---- decoder: 20 steps ----
  __bf16* dh1 = dh + (size_t)2 * kB * kHp;
  float*  dc1 = dc + (size_t)2 * kB * kH;
  for (int t = 0; t < kNY; ++t) {
    const __bf16* yt = y_bf + (size_t)t * kB * 32;
    // layer 0 (both dirs share input y_t)
    lstm_step_gemm<<<dim3(65, 1, 2), dim3(256), 0, stream>>>(yt, yt, 32, 1, dWi0_bf,
                                                             dWh0_bf, dh, db0, gates);
    dec_cell<<<dim3(cellBlocks), dim3(256), 0, stream>>>(gates, dc, dh, l0);
    // layer 1 (input = concat of layer-0 hiddens)
    lstm_step_gemm<<<dim3(65, 1, 2), dim3(256), 0, stream>>>(l0, l0, kDHK, kDHK / 32,
                                                             dWi1_bf, dWh1_bf, dh1, db1,
                                                             gates);
    dec_cell<<<dim3(cellBlocks), dim3(256), 0, stream>>>(gates, dc1, dh1, outdec);
    // attention
    gemm(outdec, kDHK, attnW1_bf, kDHK, hidPart, nullptr, kDHN, nullptr, 0, kB / 16,
         kDHN / 16, kDHK / 32, 0);
    score_k<<<dim3(kTB / 8), dim3(256), 0, stream>>>(hidPart, encPart, attn_v, score);
    softmax_k<<<dim3(kB), dim3(256), 0, stream>>>(score, aw);
    context_k<<<dim3((kB * kDHK + 255) / 256), dim3(256), 0, stream>>>(aw, outenc, ctx);
    // co = tanh([outdec, context] @ concat_W^T + concat_b)
    gemm(outdec, kDHK, concatW1_bf, kDHK, co_tmp, nullptr, kDHN, concat_b, kDH, kB / 16,
         kDHN / 16, kDHK / 32, 0);
    gemm(ctx, kDHK, concatW2_bf, kDHK, co_tmp, nullptr, kDHN, nullptr, 0, kB / 16,
         kDHN / 16, kDHK / 32, 1);
    tanhcvt_k<<<dim3((kB * kDHK + 255) / 256), dim3(256), 0, stream>>>(co_tmp, co_bf);
    // logits + CE
    gemm(co_bf, kDHK, outW_bf, kDHK, logits, nullptr, 32, out_b, kSEQ, kB / 16, 2,
         kDHK / 32, 0);
    ce_k<<<dim3(1), dim3(64), 0, stream>>>(logits, label + (size_t)t * kB, loss);
  }
}
